// AdaptiveAttention_24240795418727
// MI455X (gfx1250) — matrix operand order repaired and checked by reading, not compiled, not run on hardware
//
#include <hip/hip_runtime.h>
#include <hip/hip_bf16.h>
#include <math.h>

// CDNA5 WMMA vector types (wave32)
typedef __attribute__((ext_vector_type(16))) __bf16 v16bf;
typedef __attribute__((ext_vector_type(8)))  __bf16 v8bf;
typedef __attribute__((ext_vector_type(4)))  __bf16 v4bf;
typedef __attribute__((ext_vector_type(8)))  float  v8f;
typedef __attribute__((ext_vector_type(4)))  float  v4f;
typedef __attribute__((ext_vector_type(2)))  float  v2f;

#define N_IMG 49
#define HID   512
#define LSTR  520   // padded LDS row stride (bf16 elems): 1040 B = 4-bank shift/row

static __device__ inline v16bf cat16(v8bf lo, v8bf hi) {
    return __builtin_shufflevector(lo, hi, 0, 1, 2, 3, 4, 5, 6, 7,
                                          8, 9, 10, 11, 12, 13, 14, 15);
}

// A fragment (16x32 bf16, ISA 7.12.2): lane's 16 elems = row m=lane&15,
// k-runs [kb, kb+8) and [kb+16, kb+24), kb = kt*32 + (lane<16 ? 0 : 8).
static __device__ inline v16bf load_afrag(const __bf16* row, int kb) {
    v8bf lo = *(const v8bf*)(row + kb);
    v8bf hi = *(const v8bf*)(row + kb + 16);
    return cat16(lo, hi);
}

// B fragment (32x16 bf16): lane's 16 elems = column nc, k-runs
// [kb, kb+8) and [kb+16, kb+24), kb = kt*32 + (lane<16 ? 0 : 8), from transposed LDS.
static __device__ inline v16bf load_bfrag(const __bf16* col, int kb) {
    v8bf lo = *(const v8bf*)(col + kb);
    v8bf hi = *(const v8bf*)(col + kb + 16);
    return cat16(lo, hi);
}

// float4 -> packed 4x bf16 (compiler pairs into v_cvt_pk_bf16_f32)
static __device__ inline v4bf cvt4(v4f f) {
    v4bf h;
    h[0] = (__bf16)f[0]; h[1] = (__bf16)f[1];
    h[2] = (__bf16)f[2]; h[3] = (__bf16)f[3];
    return h;
}

// ---------------------------------------------------------------------------
// Kernel 1: sh = h_t @ W_g, rs = s_t @ W_s   (B x 512) @ (512 x 49)
// One block = 16 batch rows of ONE product (blockIdx.x & 1 picks which).
// 4 waves, wave w owns N-tile w (N padded 49->64), K=512 in 16 WMMA steps.
// ---------------------------------------------------------------------------
__global__ __launch_bounds__(128)
void aa_rowgemm_kernel(const float* __restrict__ h_t,
                       const float* __restrict__ s_t,
                       const float* __restrict__ W_g,
                       const float* __restrict__ W_s,
                       float* __restrict__ SH,
                       float* __restrict__ RS) {
    __shared__ __bf16 lds_X[16 * LSTR];   // 16.6 KB: input rows bf16 [m][k]
    __shared__ __bf16 lds_WT[64 * LSTR];  // 66.6 KB: weight TRANSPOSED [n][k]

    const int tid  = threadIdx.x;
    const int lane = tid & 31;
    const int w    = tid >> 5;

    const int matsel = blockIdx.x & 1;           // 0 -> sh, 1 -> rs
    const int b0     = (blockIdx.x >> 1) * 16;

    const float* X   = (matsel ? s_t : h_t) + (size_t)b0 * HID;
    const float* W   = matsel ? W_s : W_g;
    float*       OUT = matsel ? RS  : SH;

    // float4 global reads -> packed bf16 b64 LDS stores
    const v4f* X4 = (const v4f*)X;
    for (int c = tid; c < (16 * HID) / 4; c += 128) {
        int m  = c >> 7;                 // 128 float4 chunks per 512-row
        int kc = (c & 127) << 2;
        *(v4bf*)(lds_X + m * LSTR + kc) = cvt4(X4[c]);
    }
    for (int i = tid; i < HID * N_IMG; i += 128) {
        int d = i / N_IMG, j = i - d * N_IMG;    // W[d][j], read coalesced
        lds_WT[j * LSTR + d] = (__bf16)W[i];     // store transposed [j][d]
    }
    for (int i = tid; i < 15 * LSTR; i += 128)   // zero pad cols 49..63
        lds_WT[N_IMG * LSTR + i] = (__bf16)0.0f;
    __syncthreads();

    const int m  = lane & 15;
    const int nc = w * 16 + (lane & 15);
    const int khA = (lane < 16) ? 0 : 8;
    const int khB = (lane < 16) ? 0 : 8;
    const __bf16* arow = lds_X  + m * LSTR;
    const __bf16* bcol = lds_WT + nc * LSTR;

    v8f acc = {};
#pragma unroll 4
    for (int kt = 0; kt < 16; ++kt) {
        v16bf a  = load_afrag(arow, kt * 32 + khA);
        v16bf bf = load_bfrag(bcol, kt * 32 + khB);
        acc = __builtin_amdgcn_wmma_f32_16x16x32_bf16(
            false, a, false, bf, (short)0, acc, false, false);
    }

    if (nc < N_IMG) {
#pragma unroll
        for (int r = 0; r < 8; ++r) {            // D layout: M = r + 8*lanehalf
            int row = r + ((lane < 16) ? 0 : 8);
            OUT[(size_t)(b0 + row) * N_IMG + nc] = acc[r];
        }
    }
}

// ---------------------------------------------------------------------------
// Kernel 2: one block per batch row b. V[b] staged once to LDS (bf16) for the
// WMMA GEMM; the alpha-weighted sum re-reads V[b] from global (L2/WGP$ hit,
// fp32 precision). tanh / softmax / blend fused in-LDS.
// 8 waves: wave w -> M-tile (w>>1), two N-tiles (2*(w&1), 2*(w&1)+1).
// ---------------------------------------------------------------------------
__global__ __launch_bounds__(256)
void aa_main_kernel(const float* __restrict__ V,
                    const float* __restrict__ s_t,
                    const float* __restrict__ W_v,
                    const float* __restrict__ w_h,
                    const float* __restrict__ SH,
                    const float* __restrict__ RS,
                    float* __restrict__ out) {
    __shared__ __bf16 lds_Vbf[64 * LSTR];   // 66.6 KB: V[b] bf16 [n][k]
    __shared__ __bf16 lds_WvT[64 * LSTR];   // 66.6 KB: W_v bf16 TRANSPOSED [n][k]
    __shared__ float  lds_IP[64 * 64];      // 16.4 KB: image_part tile
    __shared__ float  lds_z[64];
    __shared__ float  lds_sh[N_IMG];
    __shared__ float  lds_rs[N_IMG];
    __shared__ float  lds_wh[N_IMG];
    __shared__ float  lds_alpha[N_IMG];

    const int b    = blockIdx.x;
    const int tid  = threadIdx.x;
    const int lane = tid & 31;
    const int w    = tid >> 5;

    const float* Vb = V + (size_t)b * N_IMG * HID;

    // V[b]: float4 coalesced HBM reads (the ONE pass of V through HBM),
    // packed bf16 b64 LDS stores.
    const v4f* Vb4 = (const v4f*)Vb;
    for (int c = tid; c < (N_IMG * HID) / 4; c += 256) {
        int n  = c >> 7;
        int kc = (c & 127) << 2;
        *(v4bf*)(lds_Vbf + n * LSTR + kc) = cvt4(Vb4[c]);
    }
    for (int i = tid; i < 15 * LSTR; i += 256)          // zero pad rows 49..63
        lds_Vbf[N_IMG * LSTR + i] = (__bf16)0.0f;
    for (int i = tid; i < HID * N_IMG; i += 256) {      // W_v coalesced (L2-hot)
        int d = i / N_IMG, j = i - d * N_IMG;
        lds_WvT[j * LSTR + d] = (__bf16)W_v[i];         // transposed [j][d]
    }
    for (int i = tid; i < 15 * LSTR; i += 256)
        lds_WvT[N_IMG * LSTR + i] = (__bf16)0.0f;
    if (tid < N_IMG) {
        lds_sh[tid] = SH[(size_t)b * N_IMG + tid];
        lds_rs[tid] = RS[(size_t)b * N_IMG + tid];
        lds_wh[tid] = w_h[tid];
    }
    __syncthreads();

    // ---- IP = V[b](49x512) @ W_v(512x49), padded to 64x64, bf16 WMMA ----
    const int mt  = w >> 1;
    const int ntb = (w & 1) * 2;
    const int n_row = mt * 16 + (lane & 15);
    const int khA = (lane < 16) ? 0 : 8;
    const int khB = (lane < 16) ? 0 : 8;
    const int nc0 = ntb * 16 + (lane & 15);
    const int nc1 = nc0 + 16;
    const __bf16* arow  = lds_Vbf + n_row * LSTR;  // rows >= 49 zeroed
    const __bf16* bcol0 = lds_WvT + nc0 * LSTR;
    const __bf16* bcol1 = lds_WvT + nc1 * LSTR;

    v8f acc0 = {}, acc1 = {};
#pragma unroll 4
    for (int kt = 0; kt < 16; ++kt) {
        v16bf a  = load_afrag(arow,  kt * 32 + khA);
        v16bf b0 = load_bfrag(bcol0, kt * 32 + khB);
        v16bf b1 = load_bfrag(bcol1, kt * 32 + khB);
        acc0 = __builtin_amdgcn_wmma_f32_16x16x32_bf16(
            false, a, false, b0, (short)0, acc0, false, false);
        acc1 = __builtin_amdgcn_wmma_f32_16x16x32_bf16(
            false, a, false, b1, (short)0, acc1, false, false);
    }
#pragma unroll
    for (int r = 0; r < 8; ++r) {
        int row = mt * 16 + r + ((lane < 16) ? 0 : 8);
        lds_IP[row * 64 + nc0] = acc0[r];
        lds_IP[row * 64 + nc1] = acc1[r];
    }
    __syncthreads();

    // ---- z[n] = sum_k tanh(IP[n,k] + sh[n]) * w_h[k]; z[49] = attn_s ----
    if (tid < N_IMG) {
        float shn = lds_sh[tid];
        float acc = 0.0f;
        for (int k = 0; k < N_IMG; ++k)
            acc += tanhf(lds_IP[tid * 64 + k] + shn) * lds_wh[k];
        lds_z[tid] = acc;
    } else if (tid == N_IMG) {
        float acc = 0.0f;
        for (int k = 0; k < N_IMG; ++k)
            acc += tanhf(lds_rs[k] + lds_sh[k]) * lds_wh[k];
        lds_z[N_IMG] = acc;
    }
    __syncthreads();

    // ---- softmaxes (49-way alpha, 50-way beta), redundant per thread ----
    float m1 = -1e30f;
    for (int n = 0; n < N_IMG; ++n) m1 = fmaxf(m1, lds_z[n]);
    float s1 = 0.0f;
    for (int n = 0; n < N_IMG; ++n) s1 += __expf(lds_z[n] - m1);
    float zs = lds_z[N_IMG];
    float m2 = fmaxf(m1, zs);
    float s2 = __expf(zs - m2);
    for (int n = 0; n < N_IMG; ++n) s2 += __expf(lds_z[n] - m2);
    float beta = __expf(zs - m2) / s2;

    if (tid < N_IMG) lds_alpha[tid] = __expf(lds_z[tid] - m1) / s1;
    __syncthreads();

    // ---- c_t from fp32 V (global, L2-hot), float2 vectorized blend ----
    const v2f* Vb2 = (const v2f*)Vb;
    const v2f* sb2 = (const v2f*)(s_t + (size_t)b * HID);
    v2f*       ob2 = (v2f*)(out + (size_t)b * HID);
    {
        const int d2 = tid;             // 256 threads x float2 = 512 columns
        float a0 = 0.0f, a1 = 0.0f;
#pragma unroll 7
        for (int n = 0; n < N_IMG; ++n) {
            v2f v = Vb2[n * (HID / 2) + d2];
            float al = lds_alpha[n];
            a0 += al * v[0];
            a1 += al * v[1];
        }
        v2f s = sb2[d2];
        v2f r;
        r[0] = beta * s[0] + (1.0f - beta) * a0;
        r[1] = beta * s[1] + (1.0f - beta) * a1;
        ob2[d2] = r;
    }
}

// ---------------------------------------------------------------------------
extern "C" void kernel_launch(void* const* d_in, const int* in_sizes, int n_in,
                              void* d_out, int out_size, void* d_ws, size_t ws_size,
                              hipStream_t stream) {
    const float* V   = (const float*)d_in[0];
    const float* s_t = (const float*)d_in[1];
    const float* h_t = (const float*)d_in[2];
    const float* W_v = (const float*)d_in[3];
    const float* W_g = (const float*)d_in[4];
    const float* W_s = (const float*)d_in[5];
    const float* w_h = (const float*)d_in[6];
    float*       out = (float*)d_out;

    const int B = in_sizes[0] / (N_IMG * HID);   // 8192

    float* SH = (float*)d_ws;                    // B x 49
    float* RS = SH + (size_t)B * N_IMG;          // B x 49

    aa_rowgemm_kernel<<<(B / 16) * 2, 128, 0, stream>>>(h_t, s_t, W_g, W_s, SH, RS);
    aa_main_kernel<<<B, 256, 0, stream>>>(V, s_t, W_v, w_h, SH, RS, out);
}
